// Elman_21638045237541
// MI455X (gfx1250) — matrix operand order repaired and checked by reading, compiled, not run on hardware
//
#include <hip/hip_runtime.h>
#include <hip/hip_bf16.h>

// ---------------------------------------------------------------------------
// Elman RNN forward for MI455X (gfx1250, wave32, WMMA, persistent scan).
// All heavy math runs through V_WMMA_F32_16X16X32_BF16 (bf16 in, f32 acc).
// ---------------------------------------------------------------------------

typedef __attribute__((ext_vector_type(16))) __bf16 v16bf;
typedef __attribute__((ext_vector_type(8)))  float  v8f;

union FragU {
    v16bf v;
    uint4 q[2];
};

// A-matrix fragment (16x32 bf16, row-major source [M,K], ld=K).
// ISA 7.12.2: lane (lr,half) holds row m=lr, K = half*8+[0..8) and 16+half*8+[0..8).
__device__ __forceinline__ v16bf load_fragA(const __bf16* __restrict__ base,
                                            int ld, int m, int k0, int half) {
    FragU f;
    const __bf16* p = base + (size_t)m * ld + k0 + half * 8;
    f.q[0] = *(const uint4*)(p);        // K = k0+half*8    .. +7
    f.q[1] = *(const uint4*)(p + 16);   // K = k0+16+half*8 .. +7
    return f.v;
}

// B-matrix fragment (32x16 bf16) from row-major W[N,K] (reference stores
// weights [out,in] == [N,K]); column n = lr, K = half*8+[0..8) and 16+half*8+[0..8).
__device__ __forceinline__ v16bf load_fragB(const __bf16* __restrict__ base,
                                            int ld, int n, int k0, int half) {
    FragU f;
    const __bf16* p = base + (size_t)n * ld + k0 + half * 8;
    f.q[0] = *(const uint4*)(p);
    f.q[1] = *(const uint4*)(p + 16);
    return f.v;
}

// f32 C/D tile: lane holds column n fixed, rows m0 + half*8 + r.
__device__ __forceinline__ void store_tile(v8f acc, const float* __restrict__ bias,
                                           float* __restrict__ Cf, __bf16* __restrict__ Cb,
                                           int m0, int n, int Nld, int half, int act) {
    float bn = bias ? bias[n] : 0.0f;
#pragma unroll
    for (int r = 0; r < 8; ++r) {
        float v = acc[r] + bn;
        if (act == 1) v = v > 0.0f ? v : 0.01f * v;   // leaky_relu slope 0.01
        size_t idx = (size_t)(m0 + half * 8 + r) * Nld + n;
        if (Cf) Cf[idx] = v;
        if (Cb) Cb[idx] = (__bf16)v;
    }
}

// ---------------------------------------------------------------------------
// GEMM: C[M,N] = act( A[M,K](bf16) * W[N,K]^T(bf16) + bias )
// 128 threads = 4 waves; wave computes a 32x64 tile: 2 A-frags x 4 B-frags ->
// 8 WMMAs per 12 b128 loads. Prefetch 256 elems (two 128-elem K-panels) ahead.
// Grid: (M/128, N/64).
// ---------------------------------------------------------------------------
__global__ void __launch_bounds__(128)
gemm_bf16_wmma(const __bf16* __restrict__ A, const __bf16* __restrict__ W,
               const float* __restrict__ bias,
               float* __restrict__ Cf, __bf16* __restrict__ Cb,
               int M, int N, int K, int act) {
    const int wave = threadIdx.x >> 5;
    const int lane = threadIdx.x & 31;
    const int half = lane >> 4;
    const int lr   = lane & 15;
    const int m0   = blockIdx.x * 128 + wave * 32;
    const int n0   = blockIdx.y * 64;

    v8f acc[2][4] = {};
    for (int kb = 0; kb < K; kb += 128) {
        // speculative prefetch of the K-panel two ahead of both streams
        __builtin_prefetch(A + (size_t)(m0 + lr) * K + kb + 256, 0, 0);
        __builtin_prefetch(W + (size_t)(n0 + lr) * K + kb + 256, 0, 0);
#pragma unroll
        for (int kk = 0; kk < 128; kk += 32) {
            const int k0 = kb + kk;
            v16bf a0 = load_fragA(A, K, m0 + lr,      k0, half);
            v16bf a1 = load_fragA(A, K, m0 + 16 + lr, k0, half);
#pragma unroll
            for (int j = 0; j < 4; ++j) {
                v16bf b = load_fragB(W, K, n0 + 16 * j + lr, k0, half);
                acc[0][j] = __builtin_amdgcn_wmma_f32_16x16x32_bf16(
                    false, a0, false, b, (short)0, acc[0][j], false, false);
                acc[1][j] = __builtin_amdgcn_wmma_f32_16x16x32_bf16(
                    false, a1, false, b, (short)0, acc[1][j], false, false);
            }
        }
    }
#pragma unroll
    for (int i = 0; i < 2; ++i)
#pragma unroll
        for (int j = 0; j < 4; ++j)
            store_tile(acc[i][j], bias, Cf, Cb, m0 + 16 * i, n0 + 16 * j + lr, N, half, act);
}

// ---------------------------------------------------------------------------
// Persistent scan kernel (one per layer): loops t = 0..T-1 internally with a
// device-wide monotonic-counter barrier (2048 graph-node launches -> 1).
// Grid = H/16 = 64 blocks x 128 threads (co-residency guaranteed: 64 WGs,
// 4 KB LDS each). Block owns 16 output columns; the 4 waves split K=H four
// ways (8-WMMA chain each) and reduce partial 16x16 tiles through LDS.
//   h_new[b,q] = tanh( (start[b,t]? 0 : h[b,:]) . Uh[q,:] + bUh[q] + xp[b,t,q] )
// Reset is folded into the A-fragment load (data zeroed, EXEC all-ones).
// h double-buffered in global; hs stored bf16 for the Wy GEMM; final carry
// written f32 straight into d_out at t == T-1.
// ---------------------------------------------------------------------------
__global__ void __launch_bounds__(128)
elman_scan_wmma(const __bf16* __restrict__ Uh, const float* __restrict__ bUh,
                const __bf16* __restrict__ xp, const unsigned char* __restrict__ start,
                __bf16* __restrict__ hA, __bf16* __restrict__ hB,
                __bf16* __restrict__ hs, float* __restrict__ hT,
                unsigned* __restrict__ bar, int T, int H) {
    __shared__ float red[4 * 256];            // [wave][m*16 + n]
    const int wave  = threadIdx.x >> 5;
    const int lane  = threadIdx.x & 31;
    const int half  = lane >> 4;
    const int lr    = lane & 15;
    const int n0    = blockIdx.x * 16;        // this block's output columns
    const int kbase = wave * (H / 4);         // this wave's K range (256)

    for (int t = 0; t < T; ++t) {
        const __bf16* h_in  = (t & 1) ? hB : hA;
        __bf16*       h_out = (t & 1) ? hA : hB;

        const bool reset = start[(size_t)lr * T + t] != 0;  // lane's A row = batch lr

        v8f acc{};
        for (int kk = 0; kk < H / 4; kk += 32) {
            const int k0 = kbase + kk;
            v16bf a{};
            if (!reset) a = load_fragA(h_in, H, lr, k0, half);
            v16bf b = load_fragB(Uh, H, n0 + lr, k0, half);
            acc = __builtin_amdgcn_wmma_f32_16x16x32_bf16(
                false, a, false, b, (short)0, acc, false, false);
        }
        // stash partial tile: lane holds column n=lr, rows half*8+r
#pragma unroll
        for (int r = 0; r < 8; ++r)
            red[wave * 256 + (half * 8 + r) * 16 + lr] = acc[r];
        __syncthreads();

        // cross-wave reduce + epilogue: 128 threads x 2 elements
#pragma unroll
        for (int e = threadIdx.x; e < 256; e += 128) {
            const int m = e >> 4;              // batch
            const int n = n0 + (e & 15);       // hidden unit
            float v = red[e] + red[256 + e] + red[512 + e] + red[768 + e];
            v = tanhf(v + bUh[n] + (float)xp[((size_t)m * T + t) * H + n]);
            const __bf16 vb = (__bf16)v;
            h_out[(size_t)m * H + n]        = vb;
            hs[((size_t)m * T + t) * H + n] = vb;
            if (t == T - 1) hT[(size_t)m * H + n] = v;   // final carry, f32
        }

        // device-wide barrier: release stores, arrive, spin, acquire
        __threadfence();
        __syncthreads();
        if (threadIdx.x == 0) {
            __hip_atomic_fetch_add(bar, 1u, __ATOMIC_ACQ_REL, __HIP_MEMORY_SCOPE_AGENT);
            const unsigned target = (unsigned)(t + 1) * (unsigned)gridDim.x;
            while (__hip_atomic_load(bar, __ATOMIC_ACQUIRE, __HIP_MEMORY_SCOPE_AGENT) < target)
                __builtin_amdgcn_s_sleep(1);
        }
        __syncthreads();
        __threadfence();                       // make peers' h_out visible here
    }
}

__global__ void f32_to_bf16_kernel(const float* __restrict__ src,
                                   __bf16* __restrict__ dst, size_t n) {
    size_t i = (size_t)blockIdx.x * blockDim.x + threadIdx.x;
    if (i < n) dst[i] = (__bf16)src[i];
}

__global__ void zero_u32_kernel(unsigned* p) { *p = 0u; }

// ---------------------------------------------------------------------------
extern "C" void kernel_launch(void* const* d_in, const int* in_sizes, int n_in,
                              void* d_out, int out_size, void* d_ws, size_t ws_size,
                              hipStream_t stream) {
    const int Bz = 16, Tz = 2048, Hz = 1024, Lz = 2;

    const float*         emb   = (const float*)d_in[0];
    const unsigned char* start = (const unsigned char*)d_in[1];   // jnp bool = 1 byte
    const float*         h0    = (const float*)d_in[2];
    const float*         Win   = (const float*)d_in[3];
    const float*         b_in  = (const float*)d_in[4];
    const float*         Wout  = (const float*)d_in[5];
    const float*         b_out = (const float*)d_in[6];
    const float*         Uh    = (const float*)d_in[7];
    const float*         bUh   = (const float*)d_in[8];
    const float*         Wh    = (const float*)d_in[9];
    const float*         Wy    = (const float*)d_in[10];
    const float*         bWy   = (const float*)d_in[11];
    const float*         Wff   = (const float*)d_in[12];
    const float*         bff   = (const float*)d_in[13];

    float* hT_out = (float*)d_out;                       // [L,B,H]
    float* y_out  = hT_out + (size_t)Lz * Bz * Hz;       // [B,T,H]

    const size_t actN = (size_t)Bz * Tz * Hz;            // 33.5M elems
    const size_t HH   = (size_t)Hz * Hz;
    __bf16* P0     = (__bf16*)d_ws;          // emb, later z
    __bf16* P1     = P0 + actN;              // layer_in (x -> ff output)
    __bf16* P2     = P1 + actN;              // xp
    __bf16* P3     = P2 + actN;              // hs
    __bf16* Win_b  = P3 + actN;
    __bf16* Wout_b = Win_b + HH;
    __bf16* Uh_b   = Wout_b + HH;
    __bf16* Wh_b   = Uh_b + (size_t)Lz * HH;
    __bf16* Wy_b   = Wh_b + (size_t)Lz * HH;
    __bf16* Wff_b  = Wy_b + (size_t)Lz * HH;
    __bf16* hA     = Wff_b + (size_t)Lz * HH;
    __bf16* hB     = hA + (size_t)Bz * Hz;
    unsigned* bar  = (unsigned*)(hB + (size_t)Bz * Hz);  // 4-byte aligned

    auto cvt = [&](const float* s, __bf16* d, size_t n) {
        f32_to_bf16_kernel<<<dim3((unsigned)((n + 255) / 256)), 256, 0, stream>>>(s, d, n);
    };

    cvt(emb,  P0,     actN);
    cvt(Win,  Win_b,  HH);
    cvt(Wout, Wout_b, HH);
    cvt(Uh,   Uh_b,   (size_t)Lz * HH);
    cvt(Wh,   Wh_b,   (size_t)Lz * HH);
    cvt(Wy,   Wy_b,   (size_t)Lz * HH);
    cvt(Wff,  Wff_b,  (size_t)Lz * HH);

    const dim3 gemmGrid((unsigned)(Bz * Tz / 128), (unsigned)(Hz / 64));  // (256,16)
    const dim3 scanGrid((unsigned)(Hz / 16));                             // 64

    // map_in: x = emb @ Win^T + b_in -> P1 (bf16)
    gemm_bf16_wmma<<<gemmGrid, 128, 0, stream>>>(P0, Win_b, b_in, nullptr, P1,
                                                 Bz * Tz, Hz, Hz, 0);

    for (int l = 0; l < Lz; ++l) {
        const __bf16* Uh_l  = Uh_b  + (size_t)l * HH;
        const __bf16* Wh_l  = Wh_b  + (size_t)l * HH;
        const __bf16* Wy_l  = Wy_b  + (size_t)l * HH;
        const __bf16* Wff_l = Wff_b + (size_t)l * HH;

        // xp = layer_in @ Wh^T (no bias) -> P2
        gemm_bf16_wmma<<<gemmGrid, 128, 0, stream>>>(P1, Wh_l, nullptr, nullptr, P2,
                                                     Bz * Tz, Hz, Hz, 0);
        // initialize carry from h0[l]; reset the device barrier counter
        cvt(h0 + (size_t)l * Bz * Hz, hA, (size_t)Bz * Hz);
        zero_u32_kernel<<<1, 1, 0, stream>>>(bar);

        // persistent scan: 1 launch for all T=2048 steps
        elman_scan_wmma<<<scanGrid, 128, 0, stream>>>(
            Uh_l, bUh + (size_t)l * Hz, P2, start, hA, hB, P3,
            hT_out + (size_t)l * Bz * Hz, bar, Tz, Hz);

        // z = hs @ Wy^T + bWy -> P0
        gemm_bf16_wmma<<<gemmGrid, 128, 0, stream>>>(P3, Wy_l, bWy + (size_t)l * Hz,
                                                     nullptr, P0, Bz * Tz, Hz, Hz, 0);
        // layer_in = leaky_relu(z @ Wff^T + bff) -> P1
        gemm_bf16_wmma<<<gemmGrid, 128, 0, stream>>>(P0, Wff_l, bff + (size_t)l * Hz,
                                                     nullptr, P1, Bz * Tz, Hz, Hz, 1);
    }

    // map_out: out = layer_in @ Wout^T + b_out -> f32 directly into d_out
    gemm_bf16_wmma<<<gemmGrid, 128, 0, stream>>>(P1, Wout_b, b_out, y_out, nullptr,
                                                 Bz * Tz, Hz, Hz, 0);
}
